// RVQTokenizer_10325101379802
// MI455X (gfx1250) — matrix operand order repaired and checked by reading, not compiled, not run on hardware
//
#include <hip/hip_runtime.h>

#define NV 262144
#define QQ 8
#define KK 1024
#define DD 64
#define WAVES 8
#define MT 2                               // M-tiles (16 rows each) per wave
#define ROWS_PER_BLOCK (WAVES * MT * 16)   // 256
#define ROWPAD 72                          // padded LDS row stride (bf16 elems), kills bank conflicts

// dynamic LDS: bf16 codebook [KK][ROWPAD] + cnorm [KK] + idxbuf [256]
#define SMEM_BYTES ((size_t)KK * ROWPAD * 2 + (size_t)KK * 4 + ROWS_PER_BLOCK * 4)

typedef __attribute__((ext_vector_type(16))) __bf16 v16bf;
typedef __attribute__((ext_vector_type(8)))  __bf16 v8bf;
typedef __attribute__((ext_vector_type(8)))  float  v8f;
typedef int v4i __attribute__((vector_size(4 * sizeof(int))));

#if defined(__has_builtin)
#  if __has_builtin(__builtin_amdgcn_global_load_async_to_lds_b128)
#    define HAVE_ASYNC_LDS 1
#  endif
#endif
#ifndef HAVE_ASYNC_LDS
#  define HAVE_ASYNC_LDS 0
#endif

// 16-byte global -> LDS copy (async on gfx1250 if available)
__device__ __forceinline__ void cp16_g2s(void* lds_dst, const void* gsrc)
{
#if HAVE_ASYNC_LDS
    __builtin_amdgcn_global_load_async_to_lds_b128(
        (__attribute__((address_space(1))) v4i*)(void*)gsrc,
        (__attribute__((address_space(3))) v4i*)lds_dst,
        0, 0);
#else
    *(float4*)lds_dst = *(const float4*)gsrc;
#endif
}

__device__ __forceinline__ void wait_async_then_barrier()
{
#if HAVE_ASYNC_LDS
    asm volatile("s_wait_asynccnt 0" ::: "memory");
#endif
    __syncthreads();
}

// ---------------------------------------------------------------------------
// Prep: fp32 codebooks -> bf16 [Q][K][D] + ||c||^2 per code.
// ---------------------------------------------------------------------------
__global__ __launch_bounds__(256) void rvq_prep(const float* __restrict__ cb,
                                                unsigned short* __restrict__ cbbf,
                                                float* __restrict__ cnorm)
{
    const int wave = threadIdx.x >> 5;
    const int lane = threadIdx.x & 31;
    const int row  = blockIdx.x * WAVES + wave;          // 0 .. QQ*KK-1
    const float* src = cb + (size_t)row * DD;

    float2 v = *(const float2*)(src + lane * 2);
    __bf16 h0 = (__bf16)v.x;
    __bf16 h1 = (__bf16)v.y;
    unsigned int pk = ((unsigned int)__builtin_bit_cast(unsigned short, h1) << 16)
                    |  (unsigned int)__builtin_bit_cast(unsigned short, h0);
    *(unsigned int*)(cbbf + (size_t)row * DD + lane * 2) = pk;

    float s = v.x * v.x + v.y * v.y;
    #pragma unroll
    for (int off = 16; off > 0; off >>= 1)
        s += __shfl_xor(s, off, 32);
    if (lane == 0) cnorm[row] = s;
}

// ---------------------------------------------------------------------------
// Main RVQ kernel. Per block: 256 rows. Per stage: async-stage the bf16
// codebook + cnorm into LDS once, then all 8 waves run 64 code tiles of
// v_wmma_f32_16x16x32_bf16 off LDS. Score = ||c||^2 - 2 r.c (C init = cnorm,
// A = bf16(-2*residual)). Residual lives in registers in A-fragment layout.
// ---------------------------------------------------------------------------
__global__ __launch_bounds__(256) void rvq_main(
    const float*          __restrict__ x,
    const float*          __restrict__ cb,     // fp32 codebooks [Q][K][D]
    const unsigned short* __restrict__ cbbf,   // bf16 codebooks [Q][K][D]
    const float*          __restrict__ cnorm,  // [Q][K]
    int*                  __restrict__ enc,    // [N][Q]
    float*                __restrict__ qout)   // [N][D]
{
    extern __shared__ unsigned char smem[];
    unsigned short* cb_s = (unsigned short*)smem;                          // [KK][ROWPAD]
    float*          cn_s = (float*)(smem + (size_t)KK * ROWPAD * 2);       // [KK]
    int*            idxb = (int*)(smem + (size_t)KK * ROWPAD * 2 + KK * 4);// [256]

    const int  tid   = threadIdx.x;
    const int  wave  = tid >> 5;
    const int  lane  = tid & 31;
    const int  li    = lane & 15;   // lane within 16-group
    const int  h     = lane >> 4;   // half-wave
    const long base  = (long)blockIdx.x * ROWS_PER_BLOCK + wave * (MT * 16);
    const int  lbase = wave * (MT * 16);

    // residual registers, A-fragment layout: r[m][chunk][elem]
    // chunk c covers dims [c*32, c*32+32): elem j<8 -> dim c*32+8h+j,
    // elem j>=8 -> dim c*32+16+8h+(j-8).
    float r[MT][2][16];
    #pragma unroll
    for (int m = 0; m < MT; ++m) {
        const float* xr = x + (size_t)(base + m * 16 + li) * DD;
        #pragma unroll
        for (int c = 0; c < 2; ++c) {
            const int d0 = c * 32 + 8 * h;
            const int d1 = c * 32 + 16 + 8 * h;
            *(float4*)&r[m][c][0]  = *(const float4*)(xr + d0);
            *(float4*)&r[m][c][4]  = *(const float4*)(xr + d0 + 4);
            *(float4*)&r[m][c][8]  = *(const float4*)(xr + d1);
            *(float4*)&r[m][c][12] = *(const float4*)(xr + d1 + 4);
        }
    }

    for (int q = 0; q < QQ; ++q) {
        // ---- stage codebook into LDS (128 KB bf16 + 4 KB cnorm) ----------
        {
            const unsigned short* cbq = cbbf + (size_t)q * KK * DD;
            #pragma unroll
            for (int i = 0; i < 32; ++i) {
                const int u   = i * 256 + tid;     // 16-byte unit id, 0..8191
                const int row = u >> 3;
                const int seg = u & 7;
                cp16_g2s(cb_s + (size_t)row * ROWPAD + seg * 8,
                         cbq + (size_t)u * 8);
            }
            cp16_g2s(cn_s + tid * 4, cnorm + (size_t)q * KK + tid * 4);
            wait_async_then_barrier();
        }

        // ---- build A fragments: bf16(-2 * residual) ----------------------
        v16bf a[MT][2];
        #pragma unroll
        for (int m = 0; m < MT; ++m)
            #pragma unroll
            for (int c = 0; c < 2; ++c)
                #pragma unroll
                for (int j = 0; j < 16; ++j)
                    a[m][c][j] = (__bf16)(-2.0f * r[m][c][j]);

        float bval[MT][8];
        int   bidx[MT][8];
        #pragma unroll
        for (int m = 0; m < MT; ++m)
            #pragma unroll
            for (int i = 0; i < 8; ++i) { bval[m][i] = 3.4028235e38f; bidx[m][i] = 0; }

        // ---- 64 code tiles off LDS ---------------------------------------
        #pragma unroll 2
        for (int t = 0; t < KK / 16; ++t) {
            const int   code = t * 16 + li;
            const float cn   = cn_s[code];
            const unsigned short* bp = cb_s + (size_t)code * ROWPAD + h * 8;
            v8bf p0 = *(const v8bf*)(bp);
            v8bf p1 = *(const v8bf*)(bp + 16);
            v8bf p2 = *(const v8bf*)(bp + 32);
            v8bf p3 = *(const v8bf*)(bp + 48);
            v16bf fb0 = __builtin_shufflevector(p0, p1, 0,1,2,3,4,5,6,7,8,9,10,11,12,13,14,15);
            v16bf fb1 = __builtin_shufflevector(p2, p3, 0,1,2,3,4,5,6,7,8,9,10,11,12,13,14,15);

            #pragma unroll
            for (int m = 0; m < MT; ++m) {
                v8f acc;
                #pragma unroll
                for (int i = 0; i < 8; ++i) acc[i] = cn;  // C = ||c||^2
                acc = __builtin_amdgcn_wmma_f32_16x16x32_bf16(
                          false, a[m][0], false, fb0, (short)0, acc, false, false);
                acc = __builtin_amdgcn_wmma_f32_16x16x32_bf16(
                          false, a[m][1], false, fb1, (short)0, acc, false, false);
                #pragma unroll
                for (int i = 0; i < 8; ++i) {
                    if (acc[i] < bval[m][i]) { bval[m][i] = acc[i]; bidx[m][i] = code; }
                }
            }
        }

        // ---- argmin across the 16 code-columns of each 16-lane group -----
        #pragma unroll
        for (int m = 0; m < MT; ++m)
            #pragma unroll
            for (int off = 1; off < 16; off <<= 1)
                #pragma unroll
                for (int i = 0; i < 8; ++i) {
                    float ov = __shfl_xor(bval[m][i], off, 16);
                    int   oi = __shfl_xor(bidx[m][i], off, 16);
                    if (ov < bval[m][i] || (ov == bval[m][i] && oi < bidx[m][i])) {
                        bval[m][i] = ov; bidx[m][i] = oi;
                    }
                }

        // ---- publish indices (lane 0 of each half owns rows i + 8h) ------
        if (li == 0) {
            #pragma unroll
            for (int m = 0; m < MT; ++m)
                #pragma unroll
                for (int i = 0; i < 8; ++i) {
                    const int rrow = m * 16 + h * 8 + i;
                    idxb[lbase + rrow] = bidx[m][i];
                    enc[(base + rrow) * QQ + q] = bidx[m][i];
                }
        }
        __syncthreads();

        // ---- residual update: r -= cb_fp32[q][idx] -----------------------
        #pragma unroll
        for (int m = 0; m < MT; ++m) {
            const int myidx = idxb[lbase + m * 16 + li];
            const float* crow = cb + ((size_t)q * KK + myidx) * DD;
            #pragma unroll
            for (int c = 0; c < 2; ++c) {
                const int d0 = c * 32 + 8 * h;
                const int d1 = c * 32 + 16 + 8 * h;
                float t16[16];
                *(float4*)&t16[0]  = *(const float4*)(crow + d0);
                *(float4*)&t16[4]  = *(const float4*)(crow + d0 + 4);
                *(float4*)&t16[8]  = *(const float4*)(crow + d1);
                *(float4*)&t16[12] = *(const float4*)(crow + d1 + 4);
                #pragma unroll
                for (int j = 0; j < 16; ++j) r[m][c][j] -= t16[j];
            }
        }
        __syncthreads();   // idxb WAR + cb_s reuse for next stage
    }

    // ---- quantized = x - final_residual ---------------------------------
    #pragma unroll
    for (int m = 0; m < MT; ++m) {
        const size_t rg   = (size_t)(base + m * 16 + li);
        const float* xr   = x + rg * DD;
        float*       orow = qout + rg * DD;
        #pragma unroll
        for (int c = 0; c < 2; ++c) {
            const int d0 = c * 32 + 8 * h;
            const int d1 = c * 32 + 16 + 8 * h;
            float xv[16], ov[16];
            *(float4*)&xv[0]  = *(const float4*)(xr + d0);
            *(float4*)&xv[4]  = *(const float4*)(xr + d0 + 4);
            *(float4*)&xv[8]  = *(const float4*)(xr + d1);
            *(float4*)&xv[12] = *(const float4*)(xr + d1 + 4);
            #pragma unroll
            for (int j = 0; j < 16; ++j) ov[j] = xv[j] - r[m][c][j];
            *(float4*)(orow + d0)     = *(float4*)&ov[0];
            *(float4*)(orow + d0 + 4) = *(float4*)&ov[4];
            *(float4*)(orow + d1)     = *(float4*)&ov[8];
            *(float4*)(orow + d1 + 4) = *(float4*)&ov[12];
        }
    }
}

extern "C" void kernel_launch(void* const* d_in, const int* in_sizes, int n_in,
                              void* d_out, int out_size, void* d_ws, size_t ws_size,
                              hipStream_t stream)
{
    const float* x  = (const float*)d_in[0];       // [N, D]
    const float* cb = (const float*)d_in[1];       // [Q, K, D]

    unsigned short* cbbf  = (unsigned short*)d_ws;                 // 1 MB bf16 codebooks
    float*          cnorm = (float*)((char*)d_ws +
                                     (size_t)QQ * KK * DD * sizeof(unsigned short));

    int*   enc  = (int*)d_out;                       // [N, Q] int32 (first output)
    float* qout = (float*)d_out + (size_t)NV * QQ;   // [N, D] fp32 (second output)

    // allow >48KB dynamic LDS (no-op if unnecessary on this runtime)
    (void)hipFuncSetAttribute((const void*)rvq_main,
                              hipFuncAttributeMaxDynamicSharedMemorySize,
                              (int)SMEM_BYTES);

    rvq_prep<<<QQ * KK / WAVES, 256, 0, stream>>>(cb, cbbf, cnorm);
    rvq_main<<<NV / ROWS_PER_BLOCK, 256, SMEM_BYTES, stream>>>(x, cb, cbbf, cnorm, enc, qout);
}
